// OuterSensorInpaintingHead_17394617548923
// MI455X (gfx1250) — matrix operand order repaired and checked by reading, not compiled, not run on hardware
//
#include <hip/hip_runtime.h>
#include <hip/hip_bf16.h>
#include <math.h>

// ---------------------------------------------------------------------------
// Problem constants (from reference)
#define BATCH 64
#define Hdim  128
#define Wdim  128
#define HW    16384          // Hdim*Wdim
#define HIDC  64
#define LATD  1024
#define C4    256            // 4*HID
#define NSEG  234
#define RR    96
#define KC    594            // 9*66 im2col depth for 3x3 conv on 66 ch
#define KCP   608            // KC padded to multiple of 32

#define DEV __device__ __forceinline__

typedef __attribute__((ext_vector_type(16))) _Float16 v16h;
typedef __attribute__((ext_vector_type(8)))  float    v8f;

DEV v8f vzero() {
  v8f z;
  #pragma unroll
  for (int i = 0; i < 8; ++i) z[i] = 0.f;
  return z;
}

DEV float gelu_f(float x) { return 0.5f * x * (1.f + erff(x * 0.70710678118654752f)); }

// ---- WMMA fragment loaders (CDNA5 wave32 16-bit layouts, ISA 7.12.2) ------
// A 16x32 f16: lanes 0-15 rows M=0..15 K={0..7,16..23}; lanes 16-31 same rows,
// K={8..15,24..31}.  B 32x16: lane = column, same K split as A (lanes 0-15
// K={0..7,16..23}, lanes 16-31 K={8..15,24..31}).  C/D f32: lane = column, VGPRs = 8 rows,
// lanes 16-31 rows 8..15.
DEV v16h frag_a_lds(const _Float16* A, int lda, int kBase, int lane) {
  v16h a; const int l = lane & 15, hi = (lane >> 4) & 1;
  const _Float16* r = A + (size_t)l * lda + kBase + hi * 8;
  #pragma unroll
  for (int i = 0; i < 8; ++i) a[i] = r[i];
  #pragma unroll
  for (int i = 0; i < 8; ++i) a[8 + i] = r[16 + i];
  return a;
}
DEV v16h frag_b_lds(const _Float16* B, int ldb, int kBase, int nBase, int lane) {
  v16h v; const int l = lane & 15, hi = (lane >> 4) & 1;
  const _Float16* c = B + (size_t)(kBase + hi * 8) * ldb + nBase + l;
  #pragma unroll
  for (int i = 0; i < 16; ++i) v[i] = c[(size_t)(i < 8 ? i : i + 8) * ldb];
  return v;
}
// Direct-from-global f32 variants (tiny GEMMs; convert on the fly)
DEV v16h frag_a_g32(const float* A, int lda, int mBase, int kBase, int lane) {
  v16h a; const int l = lane & 15, hi = (lane >> 4) & 1;
  const float* r = A + (size_t)(mBase + l) * lda + kBase + hi * 8;
  #pragma unroll
  for (int i = 0; i < 8; ++i) a[i] = (_Float16)r[i];
  #pragma unroll
  for (int i = 0; i < 8; ++i) a[8 + i] = (_Float16)r[16 + i];
  return a;
}
DEV v16h frag_b_g32(const float* B, int ldb, int kBase, int nBase, int lane) {
  v16h v; const int l = lane & 15, hi = (lane >> 4) & 1;
  const float* c = B + (size_t)(kBase + hi * 8) * ldb + nBase + l;
  #pragma unroll
  for (int i = 0; i < 16; ++i) v[i] = (_Float16)c[(size_t)(i < 8 ? i : i + 8) * ldb];
  return v;
}
DEV v8f wmma16(const v16h& a, const v16h& b, const v8f& c) {
  return __builtin_amdgcn_wmma_f32_16x16x32_f16(false, a, false, b, (short)0, c,
                                                false, false);
}

// ===========================================================================
// K1: lat[64,64] = gelu(latent[64,1024] @ W_lat[1024,64] + b_lat)
// One block, 16 waves, each owns one 16x16 tile; K=1024 in 32 steps.
__global__ void k_latent(const float* __restrict__ latent,
                         const float* __restrict__ Wl,
                         const float* __restrict__ bl,
                         float* __restrict__ lat) {
  const int tid = threadIdx.x, lane = tid & 31, wv = tid >> 5;
  const int mt = wv >> 2, nt = wv & 3;
  v8f acc = vzero();
  for (int ks = 0; ks < LATD; ks += 32) {
    v16h a = frag_a_g32(latent, LATD, mt * 16, ks, lane);
    v16h b = frag_b_g32(Wl, HIDC, ks, nt * 16, lane);
    acc = wmma16(a, b, acc);
  }
  const int col = nt * 16 + (lane & 15);
  const int row0 = mt * 16 + ((lane >> 4) & 1) * 8;
  const float bb = bl[col];
  #pragma unroll
  for (int i = 0; i < 8; ++i)
    lat[(row0 + i) * HIDC + col] = gelu_f(acc[i] + bb);
}

// ===========================================================================
// K2: 3x3 conv, Cin=66 (2 finegrid + 64 spatially-constant cond), Cout=64.
// im2col GEMM: per block M=32 pixels, N=64, K=594 (padded 608).
// The NHWC transpose + concat is folded into the LDS A-tile build.
__global__ void k_conv3x3(const float* __restrict__ fin,
                          const float* __restrict__ lat,
                          const float* __restrict__ cw,
                          const float* __restrict__ cb,
                          float* __restrict__ yout) {
  extern __shared__ char smem_raw[];
  _Float16* As = (_Float16*)smem_raw;          // [32][KCP]
  _Float16* Bs = As + 32 * KCP;                // [KCP][64]
  float* latS  = (float*)(Bs + KCP * 64);      // [64]
  const int tid = threadIdx.x;
  const int b = blockIdx.y;
  const int pix0 = blockIdx.x * 32;

  if (tid < 64) latS[tid] = lat[b * HIDC + tid];
  for (int idx = tid; idx < KCP * 64; idx += 256) {
    int k = idx >> 6, n = idx & 63;
    Bs[idx] = (k < KC) ? (_Float16)cw[k * 64 + n] : (_Float16)0.f;
  }
  __syncthreads();
  for (int idx = tid; idx < 32 * KCP; idx += 256) {
    int m = idx / KCP;
    int k = idx - m * KCP;
    float v = 0.f;
    if (k < KC) {
      int p = pix0 + m;
      int h = p >> 7, w = p & 127;
      int tap = k / 66;
      int c = k - tap * 66;
      int ty = tap / 3;
      int hh = h + ty - 1;
      int ww = w + (tap - ty * 3) - 1;
      if (hh >= 0 && hh < Hdim && ww >= 0 && ww < Wdim)
        v = (c < 2) ? fin[(((size_t)b * 2 + c) * Hdim + hh) * Wdim + ww]
                    : latS[c - 2];
    }
    As[idx] = (_Float16)v;
  }
  __syncthreads();

  const int lane = tid & 31, wv = tid >> 5;
  const int mt = wv >> 2, nt = wv & 3;         // 2 Mt x 4 Nt = 8 tiles = 8 waves
  v8f acc = vzero();
  const _Float16* At = As + mt * 16 * KCP;
  for (int ks = 0; ks < KCP; ks += 32) {
    v16h fa = frag_a_lds(At, KCP, ks, lane);
    v16h fb = frag_b_lds(Bs, 64, ks, nt * 16, lane);
    acc = wmma16(fa, fb, acc);
  }
  const int col = nt * 16 + (lane & 15);
  const int r0 = mt * 16 + ((lane >> 4) & 1) * 8;
  const float bb = cb[col];
  #pragma unroll
  for (int i = 0; i < 8; ++i) {
    size_t p = (size_t)pix0 + r0 + i;
    yout[(((size_t)b * HW) + p) * 64 + col] = acc[i] + bb;
  }
}

// ===========================================================================
// K3/K4: GroupNorm(8 groups over H,W,8ch) + exact gelu
__global__ void k_gn_stats(const float* __restrict__ y, float* __restrict__ stat) {
  __shared__ float s1[256], s2[256];
  const int g = blockIdx.x, b = blockIdx.y, tid = threadIdx.x;
  float s = 0.f, ss = 0.f;
  const float* base = y + ((size_t)b * HW) * 64 + g * 8;
  for (int p = tid; p < HW; p += 256) {
    const float4* v4 = (const float4*)(base + (size_t)p * 64);
    float4 a = v4[0], c = v4[1];
    s  += a.x + a.y + a.z + a.w + c.x + c.y + c.z + c.w;
    ss += a.x*a.x + a.y*a.y + a.z*a.z + a.w*a.w
        + c.x*c.x + c.y*c.y + c.z*c.z + c.w*c.w;
  }
  s1[tid] = s; s2[tid] = ss; __syncthreads();
  for (int o = 128; o > 0; o >>= 1) {
    if (tid < o) { s1[tid] += s1[tid + o]; s2[tid] += s2[tid + o]; }
    __syncthreads();
  }
  if (tid == 0) {
    float mu  = s1[0] / (HW * 8.f);
    float var = s2[0] / (HW * 8.f) - mu * mu;
    stat[(b * 8 + g) * 2 + 0] = mu;
    stat[(b * 8 + g) * 2 + 1] = rsqrtf(var + 1e-5f);
  }
}

__global__ void k_gn_apply(const float* __restrict__ y,
                           const float* __restrict__ stat,
                           const float* __restrict__ gs,
                           const float* __restrict__ gb,
                           float* __restrict__ xo) {
  size_t e0 = ((size_t)blockIdx.x * 256 + threadIdx.x) * 4;
  int c = (int)(e0 & 63);
  int b = (int)((e0 >> 6) >> 14);
  int g = c >> 3;
  float mu = stat[(b * 8 + g) * 2], rs = stat[(b * 8 + g) * 2 + 1];
  float4 v = *(const float4*)(y + e0);
  float4 o;
  o.x = gelu_f((v.x - mu) * rs * gs[c + 0] + gb[c + 0]);
  o.y = gelu_f((v.y - mu) * rs * gs[c + 1] + gb[c + 1]);
  o.z = gelu_f((v.z - mu) * rs * gs[c + 2] + gb[c + 2]);
  o.w = gelu_f((v.w - mu) * rs * gs[c + 3] + gb[c + 3]);
  *(float4*)(xo + e0) = o;
}

// ===========================================================================
// K5: depthwise 7x7 conv + bias + per-pixel LayerNorm.  One wave per pixel,
// each lane owns 2 channels, wave32 shfl_xor reduction for mean/var.
__global__ void k_dw_ln(const float* __restrict__ xi, const float* __restrict__ dw,
                        const float* __restrict__ db, const float* __restrict__ ls,
                        const float* __restrict__ lb, float* __restrict__ xo) {
  const int tid = threadIdx.x, lane = tid & 31, wv = tid >> 5;
  const size_t pg = (size_t)blockIdx.x * 8 + wv;       // b*HW + p
  const int b = (int)(pg >> 14);
  const int p = (int)(pg & 16383);
  const int h = p >> 7, w = p & 127;
  const int ch = lane * 2;
  float a0 = 0.f, a1 = 0.f;
  for (int ky = 0; ky < 7; ++ky) {
    int hh = h + ky - 3;
    if (hh < 0 || hh >= Hdim) continue;
    for (int kx = 0; kx < 7; ++kx) {
      int ww = w + kx - 3;
      if (ww < 0 || ww >= Wdim) continue;
      float2 v  = *(const float2*)(xi + ((((size_t)b << 14) + (hh << 7) + ww) * 64) + ch);
      float2 wt = *(const float2*)(dw + (ky * 7 + kx) * 64 + ch);
      a0 += v.x * wt.x; a1 += v.y * wt.y;
    }
  }
  a0 += db[ch]; a1 += db[ch + 1];
  float s = a0 + a1, sq = a0 * a0 + a1 * a1;
  #pragma unroll
  for (int m = 16; m >= 1; m >>= 1) {
    s  += __shfl_xor(s,  m, 32);
    sq += __shfl_xor(sq, m, 32);
  }
  float mu  = s * (1.f / 64.f);
  float var = sq * (1.f / 64.f) - mu * mu;
  float rs  = rsqrtf(var + 1e-6f);
  float2 o;
  o.x = (a0 - mu) * rs * ls[ch]     + lb[ch];
  o.y = (a1 - mu) * rs * ls[ch + 1] + lb[ch + 1];
  *(float2*)(xo + pg * 64 + ch) = o;
}

// ===========================================================================
// K6: pw1 GEMM M=B*HW, K=64, N=256, + bias + gelu, output stored f16.
// Block: M=32, N=256 -> 32 tiles, 8 waves x 4 tiles, K=2 steps.
__global__ void k_pw1(const float* __restrict__ xi, const float* __restrict__ w1,
                      const float* __restrict__ b1, _Float16* __restrict__ hO) {
  extern __shared__ char smem_raw[];
  _Float16* As = (_Float16*)smem_raw;          // [32][64]
  _Float16* Bs = As + 32 * 64;                 // [64][256]
  const int tid = threadIdx.x;
  const size_t row0 = (size_t)blockIdx.x * 32;
  for (int idx = tid; idx < 64 * 256; idx += 256) Bs[idx] = (_Float16)w1[idx];
  for (int idx = tid; idx < 32 * 64; idx += 256)
    As[idx] = (_Float16)xi[row0 * 64 + idx];
  __syncthreads();

  const int lane = tid & 31, wv = tid >> 5;
  const int n0 = (2 * wv) * 16, n1 = (2 * wv + 1) * 16;
  v8f a00 = vzero(), a01 = vzero(), a10 = vzero(), a11 = vzero();
  #pragma unroll
  for (int ks = 0; ks < 64; ks += 32) {
    v16h fa0 = frag_a_lds(As, 64, ks, lane);
    v16h fa1 = frag_a_lds(As + 16 * 64, 64, ks, lane);
    v16h fb0 = frag_b_lds(Bs, 256, ks, n0, lane);
    v16h fb1 = frag_b_lds(Bs, 256, ks, n1, lane);
    a00 = wmma16(fa0, fb0, a00);
    a10 = wmma16(fa1, fb0, a10);
    a01 = wmma16(fa0, fb1, a01);
    a11 = wmma16(fa1, fb1, a11);
  }
  const int l = lane & 15, hi = (lane >> 4) & 1;
  auto stor = [&](const v8f& ac, int mt, int nc) {
    int col = nc + l;
    int r0 = mt * 16 + hi * 8;
    float bb = b1[col];
    #pragma unroll
    for (int i = 0; i < 8; ++i)
      hO[(row0 + r0 + i) * 256 + col] = (_Float16)gelu_f(ac[i] + bb);
  };
  stor(a00, 0, n0); stor(a10, 1, n0); stor(a01, 0, n1); stor(a11, 1, n1);
}

// ===========================================================================
// K7/K8/K9: GRN  gx = sqrt(sum_hw x^2);  scale = 1 + g * gx/(mean_c gx + eps)
__global__ void k_zero(float* p, int n) {
  int i = blockIdx.x * 256 + threadIdx.x;
  if (i < n) p[i] = 0.f;
}

__global__ void k_grn_p1(const _Float16* __restrict__ hI, float* __restrict__ gsum) {
  const int b = blockIdx.y, chunk = blockIdx.x, t = threadIdx.x;  // t = channel
  const _Float16* base = hI + (((size_t)b << 14) + (size_t)chunk * 512) * 256 + t;
  float s = 0.f;
  for (int i = 0; i < 512; ++i) {
    if (i + 16 < 512) __builtin_prefetch(base + (size_t)(i + 16) * 256, 0, 0);
    float v = (float)base[(size_t)i * 256];
    s += v * v;
  }
  atomicAdd(&gsum[b * 256 + t], s);
}

__global__ void k_grn_fin(const float* __restrict__ gsum,
                          const float* __restrict__ gg,
                          float* __restrict__ scale) {
  __shared__ float sm[256];
  const int b = blockIdx.x, t = threadIdx.x;
  float g = sqrtf(gsum[b * 256 + t]);
  sm[t] = g; __syncthreads();
  for (int o = 128; o > 0; o >>= 1) {
    if (t < o) sm[t] += sm[t + o];
    __syncthreads();
  }
  float mean = sm[0] * (1.f / 256.f);
  scale[b * 256 + t] = 1.f + gg[t] * (g / (mean + 1e-6f));
}

// ===========================================================================
// K10: pw2 GEMM M=B*HW, K=256, N=64.  GRN affine (x*scale + grn_b) folded
// into the A-tile staging; residual + bias folded into epilogue.
__global__ void k_pw2(const _Float16* __restrict__ hI, const float* __restrict__ scale,
                      const float* __restrict__ gbr, const float* __restrict__ w2,
                      const float* __restrict__ b2, const float* __restrict__ res,
                      float* __restrict__ xo) {
  extern __shared__ char smem_raw[];
  _Float16* As = (_Float16*)smem_raw;          // [32][256]
  _Float16* Bs = As + 32 * 256;                // [256][64]
  const int tid = threadIdx.x;
  const size_t row0 = (size_t)blockIdx.x * 32;
  const int b = (int)(row0 >> 14);
  for (int idx = tid; idx < 256 * 64; idx += 256) Bs[idx] = (_Float16)w2[idx];
  for (int idx = tid; idx < 32 * 256; idx += 256) {
    int k = idx & 255;
    float v = (float)hI[row0 * 256 + idx];
    As[idx] = (_Float16)(v * scale[b * 256 + k] + gbr[k]);
  }
  __syncthreads();

  const int lane = tid & 31, wv = tid >> 5;
  const int mt = wv >> 2, nt = wv & 3;         // 2 Mt x 4 Nt
  v8f acc = vzero();
  const _Float16* At = As + mt * 16 * 256;
  #pragma unroll
  for (int ks = 0; ks < 256; ks += 32) {
    v16h fa = frag_a_lds(At, 256, ks, lane);
    v16h fb = frag_b_lds(Bs, 64, ks, nt * 16, lane);
    acc = wmma16(fa, fb, acc);
  }
  const int col = nt * 16 + (lane & 15);
  const int r0 = mt * 16 + ((lane >> 4) & 1) * 8;
  const float bb = b2[col];
  #pragma unroll
  for (int i = 0; i < 8; ++i) {
    size_t r = row0 + r0 + i;
    xo[r * 64 + col] = acc[i] + bb + res[r * 64 + col];
  }
}

// ===========================================================================
// K11: gather + masked softmax attention pooling.  One block per (b, ns).
__global__ void k_attn(const float* __restrict__ feats, const int* __restrict__ gidx,
                       const int* __restrict__ valid, const float* __restrict__ aw,
                       const float* __restrict__ ab, float* __restrict__ pooled) {
  __shared__ float lg[RR], at[RR], red[2];
  __shared__ int gx[RR];
  const int bi = blockIdx.x;
  const int b = bi / NSEG, ns = bi - b * NSEG;
  const int tid = threadIdx.x;
  if (tid < RR) {
    int gi = gidx[ns * RR + tid];
    gx[tid] = gi;
    const float* row = feats + (((size_t)b << 14) + gi) * 64;
    float s = 0.f;
    #pragma unroll
    for (int k = 0; k < 64; ++k) s += row[k] * aw[k];
    lg[tid] = valid[ns * RR + tid] ? (s + ab[0]) : -10000.f;
  }
  __syncthreads();
  if (tid == 0) {
    float m = lg[0];
    for (int r = 1; r < RR; ++r) m = fmaxf(m, lg[r]);
    red[0] = m;
  }
  __syncthreads();
  if (tid < RR) at[tid] = expf(lg[tid] - red[0]);
  __syncthreads();
  if (tid == 0) {
    float s = 0.f;
    for (int r = 0; r < RR; ++r) s += at[r];
    red[1] = s;
  }
  __syncthreads();
  if (tid < 64) {
    const float inv = 1.f / red[1];
    float acc = 0.f;
    for (int r = 0; r < RR; ++r)
      acc += at[r] * inv * feats[(((size_t)b << 14) + gx[r]) * 64 + tid];
    pooled[(size_t)bi * 64 + tid] = acc;
  }
}

// ===========================================================================
// K12: head.  combined = [pooled, lat] (K=128) -> gelu(@h1) -> @h2 (N=2).
// Block: 16 rows, 4 waves (one 16x16 N-tile each) for h1 via WMMA; h2 VALU.
__global__ void k_head(const float* __restrict__ pooled, const float* __restrict__ lat,
                       const float* __restrict__ w1, const float* __restrict__ b1,
                       const float* __restrict__ w2, const float* __restrict__ b2,
                       float* __restrict__ out) {
  extern __shared__ char smem_raw[];
  _Float16* Ah = (_Float16*)smem_raw;          // [16][128]
  _Float16* Bh = Ah + 16 * 128;                // [128][64]
  float* S1 = (float*)(Bh + 128 * 64);         // [16][64]
  const int tid = threadIdx.x;                 // 128 threads
  const int row0 = blockIdx.x * 16;
  for (int idx = tid; idx < 128 * 64; idx += 128) Bh[idx] = (_Float16)w1[idx];
  for (int idx = tid; idx < 16 * 128; idx += 128) {
    int m = idx >> 7, k = idx & 127;
    int row = row0 + m;
    int b = row / NSEG;
    float v = (k < 64) ? pooled[(size_t)row * 64 + k] : lat[b * 64 + (k - 64)];
    Ah[idx] = (_Float16)v;
  }
  __syncthreads();
  const int lane = tid & 31, wv = tid >> 5;
  v8f acc = vzero();
  #pragma unroll
  for (int ks = 0; ks < 128; ks += 32) {
    v16h fa = frag_a_lds(Ah, 128, ks, lane);
    v16h fb = frag_b_lds(Bh, 64, ks, wv * 16, lane);
    acc = wmma16(fa, fb, acc);
  }
  const int col = wv * 16 + (lane & 15);
  const int r0 = ((lane >> 4) & 1) * 8;
  const float bb = b1[col];
  #pragma unroll
  for (int i = 0; i < 8; ++i) S1[(r0 + i) * 64 + col] = gelu_f(acc[i] + bb);
  __syncthreads();
  if (tid < 32) {
    int m = tid >> 1, oc = tid & 1;
    float s = b2[oc];
    #pragma unroll
    for (int k = 0; k < 64; ++k) s += S1[m * 64 + k] * w2[k * 2 + oc];
    out[(size_t)(row0 + m) * 2 + oc] = s;
  }
}

// ===========================================================================
extern "C" void kernel_launch(void* const* d_in, const int* in_sizes, int n_in,
                              void* d_out, int out_size, void* d_ws, size_t ws_size,
                              hipStream_t stream) {
  (void)in_sizes; (void)n_in; (void)out_size; (void)ws_size;
  // ---- inputs (setup_inputs() dict insertion order; blk dicts in key order)
  const float* fin   = (const float*)d_in[0];   // (64,2,128,128)
  const float* ltok  = (const float*)d_in[1];   // (64,1024)
  const float* Wlat  = (const float*)d_in[2];   // (1024,64)
  const float* blat  = (const float*)d_in[3];
  const float* convw = (const float*)d_in[4];   // (3,3,66,64)
  const float* convb = (const float*)d_in[5];
  const float* gns   = (const float*)d_in[6];
  const float* gnb   = (const float*)d_in[7];
  // blk1 = d_in[8..17], blk2 = d_in[18..27]:
  //   dw_w, dw_b, ln_s, ln_b, pw1_w, pw1_b, grn_g, grn_b, pw2_w, pw2_b
  const float* attnw = (const float*)d_in[28];  // (64,1)
  const float* attnb = (const float*)d_in[29];
  const float* h1w   = (const float*)d_in[30];  // (128,64)
  const float* h1b   = (const float*)d_in[31];
  const float* h2w   = (const float*)d_in[32];  // (64,2)
  const float* h2b   = (const float*)d_in[33];
  const int*   gidx  = (const int*)d_in[34];    // (234,96) int32
  const int*   vpos  = (const int*)d_in[35];    // (234,96) bool->int

  // ---- workspace layout (~1.26 GB)
  char* ws = (char*)d_ws;
  float*    lat    = (float*)(ws + 0);           //  16 KB
  float*    stat   = (float*)(ws + 16384);       //   4 KB  (mu,rsig per b,g)
  float*    gsum   = (float*)(ws + 20480);       //  64 KB
  float*    scale  = (float*)(ws + 86016);       //  64 KB
  float*    pooled = (float*)(ws + 151552);      // 3.8 MB
  const size_t OFF0 = (size_t)4 << 20;
  const size_t SZX  = (size_t)BATCH * HW * 64 * 4;        // 256 MB
  float*    xa   = (float*)(ws + OFF0);                   // scratch activations
  float*    x0   = (float*)(ws + OFF0 + SZX);
  float*    x1   = (float*)(ws + OFF0 + 2 * SZX);
  _Float16* hbuf = (_Float16*)(ws + OFF0 + 3 * SZX);      // 512 MB (f16)

  // 1) latent embedding
  k_latent<<<1, 512, 0, stream>>>(ltok, Wlat, blat, lat);
  // 2) fused concat + 3x3 conv (im2col WMMA)  -> xa
  size_t convLds = (size_t)(32 * KCP + KCP * 64) * 2 + 64 * 4;   // 116,992 B
  k_conv3x3<<<dim3(HW / 32, BATCH), 256, convLds, stream>>>(fin, lat, convw, convb, xa);
  // 3) group norm + gelu  -> x0
  k_gn_stats<<<dim3(8, BATCH), 256, 0, stream>>>(xa, stat);
  k_gn_apply<<<(BATCH * HW * 64) / (4 * 256), 256, 0, stream>>>(xa, stat, gns, gnb, x0);

  auto convnext = [&](int p0, const float* xin, float* xout) {
    const float* dww  = (const float*)d_in[p0 + 0];
    const float* dwb  = (const float*)d_in[p0 + 1];
    const float* lns  = (const float*)d_in[p0 + 2];
    const float* lnb  = (const float*)d_in[p0 + 3];
    const float* pw1w = (const float*)d_in[p0 + 4];
    const float* pw1b = (const float*)d_in[p0 + 5];
    const float* grng = (const float*)d_in[p0 + 6];
    const float* grnb = (const float*)d_in[p0 + 7];
    const float* pw2w = (const float*)d_in[p0 + 8];
    const float* pw2b = (const float*)d_in[p0 + 9];
    k_dw_ln<<<(BATCH * HW) / 8, 256, 0, stream>>>(xin, dww, dwb, lns, lnb, xa);
    k_pw1<<<(BATCH * HW) / 32, 256, (32 * 64 + 64 * 256) * 2, stream>>>(xa, pw1w, pw1b, hbuf);
    k_zero<<<64, 256, 0, stream>>>(gsum, BATCH * 256);
    k_grn_p1<<<dim3(HW / 512, BATCH), 256, 0, stream>>>(hbuf, gsum);
    k_grn_fin<<<BATCH, 256, 0, stream>>>(gsum, grng, scale);
    k_pw2<<<(BATCH * HW) / 32, 256, (32 * 256 + 256 * 64) * 2, stream>>>(
        hbuf, scale, grnb, pw2w, pw2b, xin, xout);
  };
  // 4/5) two ConvNeXt-V2 blocks (ping-pong x0 <-> x1)
  convnext(8, x0, x1);
  convnext(18, x1, x0);

  // 6) gather + masked softmax pooling
  k_attn<<<BATCH * NSEG, 128, 0, stream>>>(x0, gidx, vpos, attnw, attnb, pooled);
  // 7) head MLP -> d_out (B,NS,2) f32
  size_t headLds = (size_t)(16 * 128 + 128 * 64) * 2 + 16 * 64 * 4;  // 24,576 B
  k_head<<<(BATCH * NSEG) / 16, 128, headLds, stream>>>(
      pooled, lat, h1w, h1b, h2w, h2b, (float*)d_out);
}
